// VQSendRecv_30468497998533
// MI455X (gfx1250) — matrix operand order repaired and checked by reading, not compiled, not run on hardware
//
#include <hip/hip_runtime.h>
#include <hip/hip_bf16.h>

typedef __bf16 bf16;
typedef __attribute__((ext_vector_type(16))) __bf16 v16bf;
typedef __attribute__((ext_vector_type(8)))  __bf16 v8bf;
typedef __attribute__((ext_vector_type(8)))  float  v8f;
typedef int v4i_vs __attribute__((vector_size(16)));   // matches builtin param type

#define NTOK 32768   // B*S
#define DIN  1024
#define DC   64
#define KCODES 4096

// LDS tile for argmin: 64 codes x 64 bf16, padded to stride 72 (bank-conflict free
// ds_load_b128: lane bank-group start = 9*lane mod 16, a permutation), double buffered
#define CT        64
#define TROW      72
#define TSTRIDE   (CT * TROW)

#if __has_builtin(__builtin_amdgcn_global_load_async_to_lds_b128) && \
    __has_builtin(__builtin_amdgcn_s_wait_asynccnt)
#define HAVE_ASYNC_LDS 1
#else
#define HAVE_ASYNC_LDS 0
#endif

// ---------------- fragment loaders (CDNA5 WMMA bf16 16x16x32 layouts) -------
// A 16x32: lane m holds row m; K chunks [8g,8g+8) and [16+8g,16+8g+8)
__device__ __forceinline__ v16bf load_frag_a_bf16(const bf16* p, int g) {
  union { v16bf v; v8bf h[2]; } u;
  u.h[0] = *(const v8bf*)(p + 8 * g);
  u.h[1] = *(const v8bf*)(p + 16 + 8 * g);
  return u.v;
}
// B 32x16: lane n holds column n; K chunks [8g,8g+8) and [16+8g,16+8g+8)
__device__ __forceinline__ v16bf load_frag_b_bf16(const bf16* p, int g) {
  union { v16bf v; v8bf h[2]; } u;
  u.h[0] = *(const v8bf*)(p + 8 * g);
  u.h[1] = *(const v8bf*)(p + 16 + 8 * g);
  return u.v;
}
// A fragment from fp32 source (convert to bf16 in-register)
__device__ __forceinline__ v16bf load_frag_a_f32(const float* p, int g) {
  float t[16];
  *(float4*)(t + 0)  = *(const float4*)(p + 8 * g);
  *(float4*)(t + 4)  = *(const float4*)(p + 8 * g + 4);
  *(float4*)(t + 8)  = *(const float4*)(p + 16 + 8 * g);
  *(float4*)(t + 12) = *(const float4*)(p + 16 + 8 * g + 4);
  v16bf a;
#pragma unroll
  for (int i = 0; i < 16; ++i) a[i] = (bf16)t[i];
  return a;
}

__device__ __forceinline__ v8f wmma_bf16(v16bf a, v16bf b, v8f c) {
  return __builtin_amdgcn_wmma_f32_16x16x32_bf16(false, a, false, b, (short)0, c,
                                                 false, false);
}

#if HAVE_ASYNC_LDS
__device__ __forceinline__ void async_copy16(const bf16* g, bf16* l) {
  __builtin_amdgcn_global_load_async_to_lds_b128(
      (__attribute__((address_space(1))) v4i_vs*)g,
      (__attribute__((address_space(3))) v4i_vs*)l, 0, 0);
}
#endif

// ---------------- prep kernels ----------------------------------------------
__global__ __launch_bounds__(256) void vq_prep_emb(const float* __restrict__ emb,
                                                   bf16* __restrict__ emb_b,
                                                   float* __restrict__ sqr,
                                                   int* __restrict__ counts) {
  int k = blockIdx.x * blockDim.x + threadIdx.x;  // 0..4095
  const float* r = emb + (size_t)k * DC;
  bf16* o = emb_b + (size_t)k * DC;
  float s = 0.f;
#pragma unroll 8
  for (int c = 0; c < DC; ++c) { float v = r[c]; s += v * v; o[c] = (bf16)v; }
  sqr[k] = s;
  counts[k] = 0;
}

// transpose+convert W_send[1024,64] -> wsend_t[64,1024] bf16 and
//                   W_recv[64,1024] -> wrecv_t[1024,64] bf16
__global__ __launch_bounds__(256) void vq_prep_w(const float* __restrict__ Wsend,
                                                 const float* __restrict__ Wrecv,
                                                 bf16* __restrict__ wsend_t,
                                                 bf16* __restrict__ wrecv_t) {
  int i = blockIdx.x * blockDim.x + threadIdx.x;  // 0..131071
  if (i < DIN * DC) {
    int c = i >> 10, k = i & (DIN - 1);
    wsend_t[i] = (bf16)Wsend[(size_t)k * DC + c];
  } else {
    int j = i - DIN * DC;
    int c = j >> 6, k = j & (DC - 1);
    wrecv_t[j] = (bf16)Wrecv[(size_t)k * DIN + c];
  }
}

// ---------------- send GEMM: z = input @ W_send + b_send --------------------
__global__ __launch_bounds__(256) void vq_send(const float* __restrict__ in,
                                               const bf16* __restrict__ wsend_t,
                                               const float* __restrict__ b_send,
                                               float* __restrict__ z_f32,
                                               bf16* __restrict__ z_b) {
  int lane = threadIdx.x & 31;
  int w = (blockIdx.x * blockDim.x + threadIdx.x) >> 5;
  int row0 = w << 4;
  int m = lane & 15, g = lane >> 4;

  v8f zero = {};
  v8f acc[4];
#pragma unroll
  for (int nt = 0; nt < 4; ++nt) acc[nt] = zero;

  const float* arow = in + (size_t)(row0 + m) * DIN;
  for (int k0 = 0; k0 < DIN; k0 += 32) {
    v16bf a = load_frag_a_f32(arow + k0, g);
#pragma unroll
    for (int nt = 0; nt < 4; ++nt) {
      int col = nt * 16 + m;
      v16bf b = load_frag_b_bf16(wsend_t + (size_t)col * DIN + k0, g);
      acc[nt] = wmma_bf16(a, b, acc[nt]);
    }
  }
#pragma unroll
  for (int nt = 0; nt < 4; ++nt) {
    int col = nt * 16 + m;
    float bb = b_send[col];
#pragma unroll
    for (int r = 0; r < 8; ++r) {
      int row = row0 + r + 8 * g;
      float v = acc[nt][r] + bb;
      z_f32[(size_t)row * DC + col] = v;
      z_b[(size_t)row * DC + col] = (bf16)v;
    }
  }
}

// ---------------- cov GEMM + fused argmin, LDS-staged codebook --------------
// block = 8 waves = 128 rows; codebook streamed through LDS in 64-code tiles
__global__ __launch_bounds__(256) void vq_argmin(const bf16* __restrict__ z_b,
                                                 const bf16* __restrict__ emb_b,
                                                 const float* __restrict__ sqr,
                                                 int* __restrict__ idx_out) {
  __shared__ bf16 tile[2 * TSTRIDE];

  int lane = threadIdx.x & 31;
  int ww = threadIdx.x >> 5;                 // wave in block
  int row0 = blockIdx.x * 128 + ww * 16;
  int m = lane & 15, g = lane >> 4;

  // copy indexing: each thread moves two 16B chunks per stage
  int row_c = threadIdx.x >> 3;              // 0..31
  int off_c = (threadIdx.x & 7) * 8;         // element offset within row

  const bf16* zr = z_b + (size_t)(row0 + m) * DC;
  v16bf a0 = load_frag_a_bf16(zr, g);
  v16bf a1 = load_frag_a_bf16(zr + 32, g);

  float best[8];
  int bidx[8];
#pragma unroll
  for (int r = 0; r < 8; ++r) { best[r] = 3.4e38f; bidx[r] = 0; }

#if HAVE_ASYNC_LDS
  // ---- async global->LDS pipeline: fill next buffer while computing -------
  {
    const bf16* src = emb_b;  // stage 0
    bf16* dst = tile;
    async_copy16(src + row_c * DC + off_c, dst + row_c * TROW + off_c);
    async_copy16(src + (row_c + 32) * DC + off_c, dst + (row_c + 32) * TROW + off_c);
  }
  for (int s = 0; s < KCODES / CT; ++s) {
    __builtin_amdgcn_s_wait_asynccnt(0);     // this thread's stage-s chunks landed
    __syncthreads();                         // everyone's landed; prev compute done
    if (s + 1 < KCODES / CT) {
      const bf16* src = emb_b + (size_t)(s + 1) * CT * DC;
      bf16* dst = tile + ((s + 1) & 1) * TSTRIDE;
      async_copy16(src + row_c * DC + off_c, dst + row_c * TROW + off_c);
      async_copy16(src + (row_c + 32) * DC + off_c, dst + (row_c + 32) * TROW + off_c);
    }
    const bf16* tb = tile + (s & 1) * TSTRIDE;
#pragma unroll
    for (int ct = 0; ct < 4; ++ct) {
      int ci = s * CT + ct * 16 + m;
      const bf16* er = tb + (ct * 16 + m) * TROW;
      v16bf b0 = load_frag_b_bf16(er, g);
      v16bf b1 = load_frag_b_bf16(er + 32, g);
      v8f acc = {};
      acc = wmma_bf16(a0, b0, acc);
      acc = wmma_bf16(a1, b1, acc);
      float sq = sqr[ci];
#pragma unroll
      for (int r = 0; r < 8; ++r) {
        float d = sq - 2.f * acc[r];
        if (d < best[r]) { best[r] = d; bidx[r] = ci; }
      }
    }
  }
#else
  // ---- fallback: register-prefetch double buffer ---------------------------
  uint4 pr0, pr1;
  {
    const bf16* src = emb_b;
    pr0 = *(const uint4*)(src + row_c * DC + off_c);
    pr1 = *(const uint4*)(src + (row_c + 32) * DC + off_c);
  }
  for (int s = 0; s < KCODES / CT; ++s) {
    bf16* dst = tile + (s & 1) * TSTRIDE;
    *(uint4*)(dst + row_c * TROW + off_c) = pr0;
    *(uint4*)(dst + (row_c + 32) * TROW + off_c) = pr1;
    if (s + 1 < KCODES / CT) {
      const bf16* src = emb_b + (size_t)(s + 1) * CT * DC;
      pr0 = *(const uint4*)(src + row_c * DC + off_c);
      pr1 = *(const uint4*)(src + (row_c + 32) * DC + off_c);
    }
    __syncthreads();
    const bf16* tb = tile + (s & 1) * TSTRIDE;
#pragma unroll
    for (int ct = 0; ct < 4; ++ct) {
      int ci = s * CT + ct * 16 + m;
      const bf16* er = tb + (ct * 16 + m) * TROW;
      v16bf b0 = load_frag_b_bf16(er, g);
      v16bf b1 = load_frag_b_bf16(er + 32, g);
      v8f acc = {};
      acc = wmma_bf16(a0, b0, acc);
      acc = wmma_bf16(a1, b1, acc);
      float sq = sqr[ci];
#pragma unroll
      for (int r = 0; r < 8; ++r) {
        float d = sq - 2.f * acc[r];
        if (d < best[r]) { best[r] = d; bidx[r] = ci; }
      }
    }
  }
#endif

  // reduce across the 16 lanes of the N-group (xor masks stay within half-wave)
#pragma unroll
  for (int r = 0; r < 8; ++r) {
    float bv = best[r];
    int bi = bidx[r];
#pragma unroll
    for (int off = 8; off >= 1; off >>= 1) {
      float ov = __shfl_xor(bv, off, 32);
      int oi = __shfl_xor(bi, off, 32);
      if (ov < bv || (ov == bv && oi < bi)) { bv = ov; bi = oi; }
    }
    if (m == 0) idx_out[row0 + r + 8 * g] = bi;
  }
}

// ---------------- recv GEMM: x = emb[idx] @ W_recv + b_recv -----------------
__global__ __launch_bounds__(256) void vq_recv(const int* __restrict__ idx,
                                               const bf16* __restrict__ emb_b,
                                               const bf16* __restrict__ wrecv_t,
                                               const float* __restrict__ b_recv,
                                               float* __restrict__ x) {
  int lane = threadIdx.x & 31;
  int w = (blockIdx.x * blockDim.x + threadIdx.x) >> 5;
  int row0 = (w >> 4) << 4;
  int c0 = (w & 15) << 6;
  int m = lane & 15, g = lane >> 4;

  int code = idx[row0 + m];
  const bf16* ar = emb_b + (size_t)code * DC;
  v16bf a0 = load_frag_a_bf16(ar, g);
  v16bf a1 = load_frag_a_bf16(ar + 32, g);

  v8f zero = {};
  v8f acc[4];
#pragma unroll
  for (int nt = 0; nt < 4; ++nt) acc[nt] = zero;

#pragma unroll
  for (int nt = 0; nt < 4; ++nt) {
    int col = c0 + nt * 16 + m;
    v16bf b0 = load_frag_b_bf16(wrecv_t + (size_t)col * DC, g);
    v16bf b1 = load_frag_b_bf16(wrecv_t + (size_t)col * DC + 32, g);
    acc[nt] = wmma_bf16(a0, b0, acc[nt]);
    acc[nt] = wmma_bf16(a1, b1, acc[nt]);
  }
#pragma unroll
  for (int nt = 0; nt < 4; ++nt) {
    int col = c0 + nt * 16 + m;
    float bb = b_recv[col];
#pragma unroll
    for (int r = 0; r < 8; ++r) {
      int row = row0 + r + 8 * g;
      x[(size_t)row * DIN + col] = acc[nt][r] + bb;
    }
  }
}

// ---------------- losses / counts / index output (fp32 exact path) ---------
__global__ __launch_bounds__(256) void vq_losses(const int* __restrict__ idx,
                                                 const float* __restrict__ z,
                                                 const float* __restrict__ emb,
                                                 int* __restrict__ counts,
                                                 float* __restrict__ lossp,
                                                 float* __restrict__ out_idx) {
  int t = blockIdx.x * blockDim.x + threadIdx.x;
  int code = idx[t];
  const float* zr = z + (size_t)t * DC;
  const float* er = emb + (size_t)code * DC;
  float s = 0.f;
#pragma unroll 8
  for (int c = 0; c < DC; ++c) { float d = er[c] - zr[c]; s += d * d; }
  atomicAdd(&counts[code], 1);
  out_idx[t] = (float)code;

  __shared__ float sm[256];
  sm[threadIdx.x] = s;
  __syncthreads();
  for (int o = 128; o > 0; o >>= 1) {
    if (threadIdx.x < o) sm[threadIdx.x] += sm[threadIdx.x + o];
    __syncthreads();
  }
  if (threadIdx.x == 0) lossp[blockIdx.x] = sm[0];
}

// ---------------- finalize: entropy + deterministic loss sum ----------------
__global__ __launch_bounds__(256) void vq_final(const int* __restrict__ counts,
                                                const float* __restrict__ lossp,
                                                float* __restrict__ out_sc) {
  __shared__ float sm[256];
  float s = 0.f;
  for (int i = threadIdx.x; i < KCODES; i += 256) {
    int c = counts[i];
    if (c > 0) { float p = (float)c / (float)NTOK; s += p * logf(p); }
  }
  sm[threadIdx.x] = s;
  __syncthreads();
  for (int o = 128; o > 0; o >>= 1) {
    if (threadIdx.x < o) sm[threadIdx.x] += sm[threadIdx.x + o];
    __syncthreads();
  }
  if (threadIdx.x == 0) {
    float loss = 0.f;
    for (int i = 0; i < NTOK / 256; ++i) loss += lossp[i];
    float ent = -sm[0] / logf(2.f);
    out_sc[0] = loss;  // embedding_loss
    out_sc[1] = loss;  // commitment_loss (identical forward value)
    out_sc[2] = ent;
  }
}

// ---------------- host launcher ---------------------------------------------
extern "C" void kernel_launch(void* const* d_in, const int* in_sizes, int n_in,
                              void* d_out, int out_size, void* d_ws, size_t ws_size,
                              hipStream_t stream) {
  (void)in_sizes; (void)n_in; (void)out_size; (void)ws_size;
  const float* input  = (const float*)d_in[0];
  const float* W_send = (const float*)d_in[1];
  const float* b_send = (const float*)d_in[2];
  const float* emb    = (const float*)d_in[3];
  const float* W_recv = (const float*)d_in[4];
  const float* b_recv = (const float*)d_in[5];

  char* ws = (char*)d_ws;
  float* z_f32    = (float*)(ws);                          // 8 MB
  bf16*  z_b      = (bf16*)(ws + 8388608);                 // 4 MB
  bf16*  emb_b    = (bf16*)(ws + 12582912);                // 512 KB
  bf16*  wsend_t  = (bf16*)(ws + 13107200);                // 128 KB
  bf16*  wrecv_t  = (bf16*)(ws + 13238272);                // 128 KB
  float* sqr      = (float*)(ws + 13369344);               // 16 KB
  int*   idx      = (int*)  (ws + 13385728);               // 128 KB
  int*   counts   = (int*)  (ws + 13516800);               // 16 KB
  float* lossp    = (float*)(ws + 13533184);               // 512 B

  float* out_x   = (float*)d_out;                          // [NTOK,1024]
  float* out_idx = out_x + (size_t)NTOK * DIN;             // [NTOK]
  float* out_sc  = out_idx + NTOK;                         // 3 scalars

  vq_prep_emb<<<KCODES / 256, 256, 0, stream>>>(emb, emb_b, sqr, counts);
  vq_prep_w<<<(2 * DIN * DC) / 256, 256, 0, stream>>>(W_send, W_recv, wsend_t, wrecv_t);
  vq_send<<<(NTOK / 16) / 8, 256, 0, stream>>>(input, wsend_t, b_send, z_f32, z_b);
  vq_argmin<<<NTOK / 128, 256, 0, stream>>>(z_b, emb_b, sqr, idx);
  vq_recv<<<((NTOK / 16) * (DIN / 64)) / 8, 256, 0, stream>>>(idx, emb_b, wrecv_t, b_recv, out_x);
  vq_losses<<<NTOK / 256, 256, 0, stream>>>(idx, z_f32, emb, counts, lossp, out_idx);
  vq_final<<<1, 256, 0, stream>>>(counts, lossp, out_sc);
}
